// SemiMarkovCRFHead_52140902973919
// MI455X (gfx1250) — matrix operand order repaired and checked by reading, not compiled, not run on hardware
//
#include <hip/hip_runtime.h>
#include <hip/hip_bf16.h>

typedef __attribute__((ext_vector_type(16))) _Float16 v16h;
typedef __attribute__((ext_vector_type(8)))  float    v8f;
typedef __attribute__((ext_vector_type(4)))  float    f32x4;
typedef __attribute__((ext_vector_type(4)))  unsigned int v4u;
typedef __attribute__((ext_vector_type(8)))  int      v8i;
typedef __attribute__((ext_vector_type(4)))  int      v4i;

#define B_ 8
#define T_ 8192
#define H_ 512
#define C_ 24
#define K_ 100
#define MROWS (B_ * T_)
#define NEGINF (-1e30f)
#define LOG2E 1.44269504088896f
#define LN2   0.6931471805599453f
#define CHUNK 32
#define NCH   (T_ / CHUNK)
#define CHELEMS (CHUNK * C_)   // 768 f32 elements per staged chunk

#if __has_include(<hip/amd_detail/amd_gfx1250_TDM.h>)
#define TDM_6ARG 1
#endif

// ---------------------------------------------------------------------------
// TDM: async DMA of one contiguous chunk of CHELEMS f32 from global to LDS.
// D# group0: count=1, lds_addr, 57-bit global addr, type=2.
// D# group1: data_size=2 (4B), tensor_dim0=tile_dim0=CHELEMS (1-D tile),
//            tensor_dim1=tile_dim1=1, stride0=CHELEMS.
// ---------------------------------------------------------------------------
__device__ __forceinline__ void tdm_load_chunk(const float* gptr, unsigned int lds_byte) {
  unsigned long long ga = (unsigned long long)(uintptr_t)gptr;
  v4u g0 = { 1u,                                   // count=1, user mode
             lds_byte,                             // lds_addr
             (unsigned int)ga,                     // global_addr[31:0]
             (unsigned int)((ga >> 32) & 0x01FFFFFFu) | (2u << 30) }; // ga[56:32] | type=2
  v8i g1 = { (int)(2u << 16),                      // wg_mask=0, data_size=2 (4B)
             (int)(((unsigned)CHELEMS & 0xFFFFu) << 16),  // tensor_dim0[15:0] @ [63:48]
             (int)((((unsigned)CHELEMS >> 16) & 0xFFFFu) | (1u << 16)), // dim0 hi | tensor_dim1=1
             (int)((unsigned)CHELEMS << 16),       // tensor_dim1 hi=0 | tile_dim0=CHELEMS
             1,                                    // tile_dim1=1, tile_dim2=0
             (int)CHELEMS,                         // tensor_dim0_stride[31:0]
             0, 0 };
  v4i z4 = { 0, 0, 0, 0 };
#ifdef TDM_6ARG
  v8i z8 = { 0, 0, 0, 0, 0, 0, 0, 0 };
  __builtin_amdgcn_tensor_load_to_lds(g0, g1, z4, z4, z8, 0);
#else
  __builtin_amdgcn_tensor_load_to_lds(g0, g1, z4, z4, 0);
#endif
}

// ---------------------------------------------------------------------------
// Kernel 1: scores = hidden_states @ W_proj + b_proj
// f32 inputs via split-f16 WMMA: x = hi + lo, A@B ~= Ah@Bh + Ah@Bl + Al@Bh
// ---------------------------------------------------------------------------
__global__ __launch_bounds__(128) void proj_wmma(const float* __restrict__ hs,
                                                 const float* __restrict__ W,
                                                 const float* __restrict__ bias,
                                                 float* __restrict__ scores) {
  const int lane  = threadIdx.x & 31;
  const int wid   = threadIdx.x >> 5;
  const int tile  = blockIdx.x * 4 + wid;
  const int m_tile = tile >> 1;
  const int n_tile = tile & 1;
  const int ml    = lane & 15;
  const int khalf = lane >> 4;

  const int row = m_tile * 16 + ml;
  const int col = n_tile * 16 + ml;
  const bool colok = (col < C_);
  const float* rowp = hs + (size_t)row * H_;

  v8f acc = {};
  for (int k0 = 0; k0 < H_; k0 += 32) {
    const int b1 = k0 + khalf * 8;
    const int b2 = k0 + 16 + khalf * 8;
    f32x4 a0 = *(const f32x4*)(rowp + b1);
    f32x4 a1 = *(const f32x4*)(rowp + b1 + 4);
    f32x4 a2 = *(const f32x4*)(rowp + b2);
    f32x4 a3 = *(const f32x4*)(rowp + b2 + 4);
    float af[16];
#pragma unroll
    for (int j = 0; j < 4; ++j) {
      af[j] = a0[j]; af[4 + j] = a1[j]; af[8 + j] = a2[j]; af[12 + j] = a3[j];
    }
    float bf[16];
#pragma unroll
    for (int j = 0; j < 16; ++j) {
      int kk = k0 + khalf * 8 + j + (j >= 8 ? 8 : 0);
      bf[j] = colok ? W[(size_t)kk * C_ + col] : 0.0f;
    }
    v16h ah, al, bh, bl;
#pragma unroll
    for (int j = 0; j < 16; ++j) {
      _Float16 h = (_Float16)af[j];
      ah[j] = h;
      al[j] = (_Float16)(af[j] - (float)h);
      _Float16 g = (_Float16)bf[j];
      bh[j] = g;
      bl[j] = (_Float16)(bf[j] - (float)g);
    }
    acc = __builtin_amdgcn_wmma_f32_16x16x32_f16(false, ah, false, bh, (short)0, acc, false, false);
    acc = __builtin_amdgcn_wmma_f32_16x16x32_f16(false, ah, false, bl, (short)0, acc, false, false);
    acc = __builtin_amdgcn_wmma_f32_16x16x32_f16(false, al, false, bh, (short)0, acc, false, false);
  }

  if (colok) {
    float bv = bias[col];
#pragma unroll
    for (int i = 0; i < 8; ++i) {
      int m = m_tile * 16 + i + 8 * khalf;
      scores[(size_t)m * C_ + col] = acc[i] + bv;
    }
  }
}

// ---------------------------------------------------------------------------
// Kernel 2: streaming semi-CRF scan, one block per batch.
//   u[s,c] = LSE_{c'}(alpha[s,c'] + trans[c',c]) - cs[s,c]   (K-row LDS ring)
//   alpha[t,c] = cs[t,c] + LSE_{k=1..K}(dur_bias[k-1,c] + u[t-k,c])
// Scores are prefetched by TDM into a double-buffered LDS stage.
// ---------------------------------------------------------------------------
__global__ __launch_bounds__(128) void semicrf_scan(const float* __restrict__ scores,
                                                    const float* __restrict__ trans,
                                                    const float* __restrict__ db,
                                                    const int* __restrict__ lengths,
                                                    float* __restrict__ out) {
  __shared__ float u_lds[K_ * C_];
  __shared__ float db_lds[K_ * C_];
  __shared__ float tr_lds[C_ * C_];
  __shared__ float alpha_lds[C_];
  __shared__ float saved_lds[C_];
  __shared__ float pmx[4 * C_];
  __shared__ float psum[4 * C_];
  __shared__ float stage[2][CHELEMS];

  const int b = blockIdx.x;
  const int tid = threadIdx.x;
  const int c = tid % C_;
  const int slice = tid / C_;
  const int lenb = lengths[b];
  const float* sc = scores + (size_t)b * T_ * C_;
  const unsigned int stage_base = (unsigned int)(uintptr_t)(void*)&stage[0][0];

  // Kick off DMA of chunk 0 while we set up constants.
  tdm_load_chunk(sc, stage_base);

  for (int i = tid; i < K_ * C_; i += 128) db_lds[i] = db[i];
  for (int i = tid; i < C_ * C_; i += 128) tr_lds[i] = trans[i];
  if (tid < C_) saved_lds[tid] = NEGINF;
  __syncthreads();

  // Hoist time-invariant LDS data into registers.
  float dbv[25];
  if (tid < 4 * C_) {
#pragma unroll
    for (int j = 0; j < 25; ++j) dbv[j] = db_lds[(slice * 25 + j) * C_ + c];
  }
  float trc[C_];
  if (tid < C_) {
#pragma unroll
    for (int cp = 0; cp < C_; ++cp) trc[cp] = tr_lds[cp * C_ + c];
    // u[0,c] = LSE_{c'} trans[c',c]   (alpha[0]=0, cs[0]=0)
    float mx = NEGINF;
#pragma unroll
    for (int cp = 0; cp < C_; ++cp) mx = fmaxf(mx, trc[cp]);
    float s = 0.0f;
#pragma unroll
    for (int cp = 0; cp < C_; ++cp) s += __builtin_exp2f((trc[cp] - mx) * LOG2E);
    u_lds[c] = mx + __builtin_log2f(s) * LN2;
  }
  __syncthreads();

  float cs_c = 0.0f;  // running cumsum for label c (wave-0 lanes)

  for (int t = 1; t <= T_; ++t) {
    const int tm1 = t - 1;
    if ((tm1 & (CHUNK - 1)) == 0) {
      // Chunk boundary: prefetch chunk ch+1, wait for chunk ch.
      const int ch = tm1 >> 5;
      if (ch + 1 < NCH) {
        tdm_load_chunk(sc + (size_t)(ch + 1) * CHELEMS,
                       stage_base + (unsigned int)(((ch + 1) & 1) * CHELEMS * 4));
        __builtin_amdgcn_s_wait_tensorcnt(1);   // oldest (chunk ch) complete
      } else {
        __builtin_amdgcn_s_wait_tensorcnt(0);
      }
    }

    // ---- Phase A: partial LSE over this thread's 25 durations ----
    if (tid < 4 * C_) {
      const int kk = slice * 25;      // (k-1) base
      const int s0 = tm1 - kk;        // segment start for j=0
      int slot0 = s0 % K_;
      if (slot0 < 0) slot0 += K_;
      float v[25];
#pragma unroll
      for (int j = 0; j < 25; ++j) {
        int sj = slot0 - j;
        if (sj < 0) sj += K_;
        v[j] = ((s0 - j) >= 0) ? (dbv[j] + u_lds[sj * C_ + c]) : NEGINF;
      }
      float mx = NEGINF;
#pragma unroll
      for (int j = 0; j < 25; ++j) mx = fmaxf(mx, v[j]);
      float s = 0.0f;
#pragma unroll
      for (int j = 0; j < 25; ++j) s += __builtin_exp2f((v[j] - mx) * LOG2E);
      pmx[slice * C_ + c] = mx;
      psum[slice * C_ + c] = s;
    }
    __syncthreads();

    // ---- Phase B + C: wave 0 only (in-wave LDS ordering, no barrier) ----
    if (tid < C_) {
      cs_c += stage[(tm1 >> 5) & 1][(tm1 & (CHUNK - 1)) * C_ + c];
      float M = pmx[c];
      M = fmaxf(M, pmx[C_ + c]);
      M = fmaxf(M, pmx[2 * C_ + c]);
      M = fmaxf(M, pmx[3 * C_ + c]);
      float S = 0.0f;
#pragma unroll
      for (int i = 0; i < 4; ++i)
        S += psum[i * C_ + c] * __builtin_exp2f((pmx[i * C_ + c] - M) * LOG2E);
      float alpha = cs_c + M + __builtin_log2f(S) * LN2;
      alpha_lds[c] = alpha;
      if (t == lenb) saved_lds[c] = alpha;

      // Phase C: u[t,c] = LSE_{c'}(alpha[c'] + trans[c',c]) - cs[t,c]
      float w[C_];
#pragma unroll
      for (int cp = 0; cp < C_; ++cp) w[cp] = alpha_lds[cp] + trc[cp];
      float mmx = NEGINF;
#pragma unroll
      for (int cp = 0; cp < C_; ++cp) mmx = fmaxf(mmx, w[cp]);
      float ss = 0.0f;
#pragma unroll
      for (int cp = 0; cp < C_; ++cp) ss += __builtin_exp2f((w[cp] - mmx) * LOG2E);
      u_lds[(t % K_) * C_ + c] = (mmx + __builtin_log2f(ss) * LN2) - cs_c;
    }
    __syncthreads();
  }

  if (tid == 0) {
    float mx = NEGINF;
    for (int cp = 0; cp < C_; ++cp) mx = fmaxf(mx, saved_lds[cp]);
    float s = 0.0f;
    for (int cp = 0; cp < C_; ++cp) s += __builtin_exp2f((saved_lds[cp] - mx) * LOG2E);
    out[b] = mx + __builtin_log2f(s) * LN2;
  }
}

// ---------------------------------------------------------------------------
extern "C" void kernel_launch(void* const* d_in, const int* in_sizes, int n_in,
                              void* d_out, int out_size, void* d_ws, size_t ws_size,
                              hipStream_t stream) {
  const float* hs    = (const float*)d_in[0];   // (B,T,H) f32
  const float* W     = (const float*)d_in[1];   // (H,C)  f32
  const float* bias  = (const float*)d_in[2];   // (C,)   f32
  const float* trans = (const float*)d_in[3];   // (C,C)  f32
  const float* db    = (const float*)d_in[4];   // (K,C)  f32
  const int*   len   = (const int*)d_in[5];     // (B,)   i32
  float* out = (float*)d_out;                   // (B,)   f32
  float* scores = (float*)d_ws;                 // (B*T,C) f32 scratch (6.3 MB)

  const int n_tiles = (MROWS / 16) * 2;
  proj_wmma<<<dim3(n_tiles / 4), dim3(128), 0, stream>>>(hs, W, bias, scores);
  semicrf_scan<<<dim3(B_), dim3(128), 0, stream>>>(scores, trans, db, len, out);
}
